// LinearAttention_63290638074526
// MI455X (gfx1250) — matrix operand order repaired and checked by reading, not compiled, not run on hardware
//
#include <hip/hip_runtime.h>
#include <hip/hip_bf16.h>
#include <math.h>

typedef __attribute__((ext_vector_type(16))) _Float16 v16h;
typedef __attribute__((ext_vector_type(2)))  _Float16 v2h;
typedef __attribute__((ext_vector_type(8)))  float    v8f;

#define NSPAT   32768          // 32*32*32
#define DIMC    256
#define INNERC  512
#define HEADS   8
#define RM      512            // R^3 reduced tokens
#define SCALEQ  0.125f         // 64^-0.5

// ---------------- workspace layout (bytes) ----------------
#define WS_XRED   0
#define WS_KT     (512u*1024u)
#define WS_V      (1024u*1024u)
#define WS_WQ     (1536u*1024u)
#define WS_WOUT   (1792u*1024u)
#define WS_O16    (2048u*1024u)

__device__ __forceinline__ v8f wmma16(const v16h a, const v16h b, v8f c) {
  return __builtin_amdgcn_wmma_f32_16x16x32_f16(false, a, false, b, (short)0, c, false, false);
}

__device__ __forceinline__ void interp8(int o, int& lo, float& fr) {
  float p = (float)o * (31.0f / 7.0f);
  int l = (int)floorf(p);
  if (l > 30) l = 30;
  if (l < 0)  l = 0;
  lo = l;
  fr = p - (float)l;
}

// ---------- kernel 1: x_red = reduce3(x)  [256][512] ----------
__global__ void xred_kernel(const float* __restrict__ x, float* __restrict__ xred) {
  const int s = blockIdx.x;        // 0..511 output spatial
  const int c = threadIdx.x;       // channel
  const int i = s >> 6, j = (s >> 3) & 7, k = s & 7;
  int ld, lh, lw; float fd, fh, fw;
  interp8(i, ld, fd); interp8(j, lh, fh); interp8(k, lw, fw);
  const float* p0 = x + (size_t)c * NSPAT + ld * 1024 + lh * 32 + lw;
  float c00 = p0[0]       * (1.f - fw) + p0[1]       * fw;
  float c01 = p0[32]      * (1.f - fw) + p0[33]      * fw;
  float c10 = p0[1024]    * (1.f - fw) + p0[1025]    * fw;
  float c11 = p0[1024+32] * (1.f - fw) + p0[1024+33] * fw;
  float c0 = c00 * (1.f - fh) + c01 * fh;
  float c1 = c10 * (1.f - fh) + c11 * fh;
  xred[c * RM + s] = c0 * (1.f - fd) + c1 * fd;
}

// ---------- kernel 2: k,v in WMMA B-fragment layout (f16) ----------
// kT_frag[h][kc(2)][mt(32)][lane(32)][e(16)] : B for S = q*kT, K=dhead, N=m
// v_frag [h][kc(16)][nt(4)][lane(32)][e(16)] : B for o = p*v,  K=m,     N=dd
// Each 256-thread block needs exactly 16 consecutive x_red columns and 16
// w_qkv rows (stride 8) -> stage both in LDS, dot is LDS-only.
__global__ void fragkv_kernel(const float* __restrict__ w_qkv,
                              const float* __restrict__ xred,
                              _Float16* __restrict__ kT_frag,
                              _Float16* __restrict__ v_frag) {
  __shared__ float xcol[256][16];     // [cin][m-m_base]
  __shared__ float wrow[16][260];     // [row][cin], padded vs bank conflicts
  const int t = threadIdx.x;
  const int tid = blockIdx.x * 256 + t;
  const int e = tid & 15;
  const int lane = (tid >> 4) & 31;   // fragment-layout lane (block-constant hi bit)
  const int lrow = lane & 15;
  const int khalf = (lane >> 4) * 16;
  int c_base, m_base, widx, xidx;
  _Float16* dst;
  if (tid < 262144) {                       // kT fragments
    const int mt = (tid >> 9) & 31;
    const int kc = (tid >> 14) & 1;
    const int h  = tid >> 15;
    m_base = mt * 16;                       // + lrow
    c_base = 512 + (kc * 32 + khalf) * 8 + h;   // + 8*e
    widx = e; xidx = lrow;
    dst = kT_frag + tid;
  } else {                                  // v fragments
    const int i2 = tid - 262144;
    const int nt = (i2 >> 9) & 3;
    const int kc = (i2 >> 11) & 15;
    const int h  = i2 >> 15;
    m_base = kc * 32 + khalf;               // + e
    c_base = 1024 + nt * 128 + h;           // + 8*lrow
    widx = lrow; xidx = e;
    dst = v_frag + i2;
  }
  for (int i = t; i < 4096; i += 256)
    xcol[i >> 4][i & 15] = xred[(i >> 4) * RM + m_base + (i & 15)];
  for (int i = t; i < 4096; i += 256)
    wrow[i >> 8][i & 255] = w_qkv[(size_t)(c_base + (i >> 8) * 8) * DIMC + (i & 255)];
  __syncthreads();
  float acc = 0.f;
  for (int ci = 0; ci < 256; ci += 4) {
    const float4 w4 = *(const float4*)&wrow[widx][ci];
    acc += w4.x * xcol[ci][xidx] + w4.y * xcol[ci + 1][xidx]
         + w4.z * xcol[ci + 2][xidx] + w4.w * xcol[ci + 3][xidx];
  }
  *dst = (_Float16)acc;
}

// ---------- kernel 3: weight fragments ----------
__global__ void fragw_kernel(const float* __restrict__ w_qkv,
                             const float* __restrict__ w_out,
                             _Float16* __restrict__ wq_frag,
                             _Float16* __restrict__ wout_frag) {
  const int tid = blockIdx.x * 256 + threadIdx.x;   // 0..262143
  const int e = tid & 15;
  const int lane = (tid >> 4) & 31;
  const int lrow = lane & 15;
  if (tid < 131072) {                       // wq (B layout): K=cin, N=dd
    const int nt = (tid >> 9) & 3;
    const int kc = (tid >> 11) & 7;
    const int h  = tid >> 14;
    const int dd = nt * 16 + lrow;
    const int cin = kc * 32 + ((lane >> 4) * 16) + e;
    const int c = dd * HEADS + h;
    wq_frag[tid] = (_Float16)w_qkv[(size_t)c * DIMC + cin];
  } else {                                  // wout (A layout): M=oc, K=c
    const int i2 = tid - 131072;
    const int kc = (i2 >> 9) & 15;
    const int oct = i2 >> 13;
    const int oc = oct * 16 + lrow;
    const int sel = (lane >> 4) * 8;
    const int cK = kc * 32 + sel * 2 + e;
    wout_frag[i2] = (_Float16)w_out[(size_t)oc * INNERC + cK];
  }
}

// ---------- kernel 4: fused attention ----------
// block = 128 threads (4 waves); grid = (512 n-tiles of 64, 8 heads)
// LDS (aliased lifetimes):
//   [0, 32K)        x16[256][64] f16      (q phase only)
//   [0, 132096)     Sb[64][516] f32       (score phase onward)
//   [132096, +9216) qbuf[64][72] f16      (q -> score frag build)
//   [132096,+16896) obuf[64][66] f32      (o phase onward)
#define SB_STRIDE 516
#define QB_STRIDE 72
#define OB_STRIDE 66
#define SM_QOB    132096
#define SM_ATTN   (132096 + 64 * OB_STRIDE * 4)   // 148992 bytes -> 2 blocks/WGP

__global__ void __launch_bounds__(128, 1)
attn_kernel(const float* __restrict__ x,
            const _Float16* __restrict__ kT_frag,
            const _Float16* __restrict__ v_frag,
            const _Float16* __restrict__ wq_frag,
            float* __restrict__ attn_out,
            _Float16* __restrict__ o16) {
  extern __shared__ char smem[];
  _Float16* x16  = (_Float16*)(smem);           // q phase only
  float*    Sb   = (float*)(smem);              // score phase onward
  _Float16* qbuf = (_Float16*)(smem + SM_QOB);
  float*    obuf = (float*)(smem + SM_QOB);

  const int tid  = threadIdx.x;
  const int lane = tid & 31;
  const int w    = tid >> 5;           // wave id 0..3
  const int h    = blockIdx.y;
  const int n0   = blockIdx.x * 64;

  const int row = lane & 15;           // A-frag row / B,D column
  const int hi  = lane >> 4;
  const int sel = hi * 8;

  // stage x tile as f16 (packed pairs)
  for (int i = tid; i < 256 * 32; i += 128) {
    const int c = i >> 5, np = (i & 31) * 2;
    const float2 f = *(const float2*)(x + (size_t)c * NSPAT + n0 + np);
    v2h p; p[0] = (_Float16)f.x; p[1] = (_Float16)f.y;
    *(v2h*)(x16 + c * 64 + np) = p;
  }
  __syncthreads();

  v8f z = {};

  // ---- Q = x^T * Wq_h : A-frag built once per kc, reused over 4 dd-tiles ----
  {
    v8f qacc[4];
#pragma unroll
    for (int d = 0; d < 4; ++d) qacc[d] = z;
    for (int kc = 0; kc < 8; ++kc) {
      v16h a;
#pragma unroll
      for (int e = 0; e < 16; ++e) {
        const int koff = sel * 2 + e;
        a[e] = x16[(kc * 32 + koff) * 64 + w * 16 + row];
      }
#pragma unroll
      for (int ddt = 0; ddt < 4; ++ddt) {
        const v16h b = *(const v16h*)(wq_frag + ((((size_t)h * 8 + kc) * 4 + ddt) * 32 + lane) * 16);
        qacc[ddt] = wmma16(a, b, qacc[ddt]);
      }
    }
#pragma unroll
    for (int ddt = 0; ddt < 4; ++ddt)
#pragma unroll
      for (int r = 0; r < 8; ++r)
        qbuf[(w * 16 + r + hi * 8) * QB_STRIDE + ddt * 16 + row] = (_Float16)(qacc[ddt][r] * SCALEQ);
  }
  __syncthreads();   // closes x16 reads; Sb writes follow

  // ---- S = q * kT : q A-frags hoisted out of the 32-m-tile loop ----
  {
    v16h a0, a1;
#pragma unroll
    for (int e = 0; e < 16; ++e) {
      const int koff = sel * 2 + e;
      a0[e] = qbuf[(w * 16 + row) * QB_STRIDE + koff];
      a1[e] = qbuf[(w * 16 + row) * QB_STRIDE + 32 + koff];
    }
    for (int mt = 0; mt < 32; ++mt) {
      v8f acc = z;
      const v16h b0 = *(const v16h*)(kT_frag + ((((size_t)h * 2 + 0) * 32 + mt) * 32 + lane) * 16);
      const v16h b1 = *(const v16h*)(kT_frag + ((((size_t)h * 2 + 1) * 32 + mt) * 32 + lane) * 16);
      acc = wmma16(a0, b0, acc);
      acc = wmma16(a1, b1, acc);
#pragma unroll
      for (int r = 0; r < 8; ++r)
        Sb[(w * 16 + r + hi * 8) * SB_STRIDE + mt * 16 + row] = acc[r];
    }
  }
  __syncthreads();

  // ---- softmax stats (float4 scans; lane owns row=lane%16, half hi) ----
  float inv;
  {
    float4* S4 = (float4*)(Sb + (w * 16 + row) * SB_STRIDE + hi * 256);
    float mx = -3.4e38f;
    for (int i = 0; i < 64; ++i) {
      const float4 f = S4[i];
      mx = fmaxf(mx, fmaxf(fmaxf(f.x, f.y), fmaxf(f.z, f.w)));
    }
    mx = fmaxf(mx, __shfl_xor(mx, 16));
    float sum = 0.f;
    for (int i = 0; i < 64; ++i) {
      float4 f = S4[i];
      f.x = __expf(f.x - mx); f.y = __expf(f.y - mx);
      f.z = __expf(f.z - mx); f.w = __expf(f.w - mx);
      S4[i] = f;
      sum += (f.x + f.y) + (f.z + f.w);
    }
    sum += __shfl_xor(sum, 16);
    inv = 1.0f / sum;
  }
  __syncthreads();

  // ---- write normalized attn (fp32, coalesced float4) ----
  for (int r = 0; r < 16; ++r) {
    const float invr = __shfl(inv, r);
    const float4* Sr = (const float4*)(Sb + (w * 16 + r) * SB_STRIDE);
    float4* dst = (float4*)(attn_out + ((size_t)h * NSPAT + n0 + w * 16 + r) * 512);
#pragma unroll
    for (int i = 0; i < 4; ++i) {
      float4 v = Sr[lane + 32 * i];
      v.x *= invr; v.y *= invr; v.z *= invr; v.w *= invr;
      dst[lane + 32 * i] = v;
    }
  }

  // ---- o = p * v : p A-frag built once per kc, reused over 4 dd-tiles ----
  {
    v8f oacc[4];
#pragma unroll
    for (int d = 0; d < 4; ++d) oacc[d] = z;
    for (int kc = 0; kc < 16; ++kc) {
      v16h a;
#pragma unroll
      for (int e = 0; e < 16; ++e) {
        const int koff = sel * 2 + e;
        a[e] = (_Float16)(Sb[(w * 16 + row) * SB_STRIDE + kc * 32 + koff] * inv);
      }
#pragma unroll
      for (int ddt = 0; ddt < 4; ++ddt) {
        const v16h b = *(const v16h*)(v_frag + ((((size_t)h * 16 + kc) * 4 + ddt) * 32 + lane) * 16);
        oacc[ddt] = wmma16(a, b, oacc[ddt]);
      }
    }
    __syncthreads();   // qbuf reads long done; obuf overwrites that region
#pragma unroll
    for (int ddt = 0; ddt < 4; ++ddt)
#pragma unroll
      for (int r = 0; r < 8; ++r)
        obuf[(w * 16 + r + hi * 8) * OB_STRIDE + ddt * 16 + row] = oacc[ddt][r];
  }
  __syncthreads();

  // ---- store o16[c][n] packed, c = dd*8 + h ----
  for (int i = tid; i < 64 * 32; i += 128) {
    const int dd = i >> 5, nn = (i & 31) * 2;
    v2h p;
    p[0] = (_Float16)obuf[nn * OB_STRIDE + dd];
    p[1] = (_Float16)obuf[(nn + 1) * OB_STRIDE + dd];
    *(v2h*)(o16 + (size_t)(dd * HEADS + h) * NSPAT + n0 + nn) = p;
  }
}

// ---------- kernel 5: out = Wout * o ----------
// 4 waves; wave w -> oc-tiles {4w..4w+3} x all 4 n-tiles: each w_out A-frag
// load feeds 4 WMMAs (4x less L2 traffic than 1 n-tile/wave).
#define OL_STRIDE 68
__global__ void __launch_bounds__(128, 1)
outgemm_kernel(const _Float16* __restrict__ o16,
               const _Float16* __restrict__ wout_frag,
               float* __restrict__ out) {
  extern __shared__ char smem[];
  _Float16* ol = (_Float16*)smem;     // [512][OL_STRIDE]
  const int tid = threadIdx.x, lane = tid & 31, w = tid >> 5;
  const int n0 = blockIdx.x * 64;
  for (int i = tid; i < 512 * 16; i += 128) {   // 16 uint2 per channel row
    const int c = i >> 4, p = i & 15;
    const uint2 val = *(const uint2*)(o16 + (size_t)c * NSPAT + n0 + p * 4);
    *(uint2*)(ol + c * OL_STRIDE + p * 4) = val;
  }
  __syncthreads();

  const int col = lane & 15, hi = lane >> 4;
  v8f acc[4][4];
  v8f z = {};
#pragma unroll
  for (int a4 = 0; a4 < 4; ++a4)
#pragma unroll
    for (int nt = 0; nt < 4; ++nt) acc[a4][nt] = z;

  for (int kc = 0; kc < 16; ++kc) {
    v16h bfrag[4];
#pragma unroll
    for (int nt = 0; nt < 4; ++nt)
#pragma unroll
      for (int e = 0; e < 16; ++e)
        bfrag[nt][e] = ol[(kc * 32 + hi * 16 + e) * OL_STRIDE + nt * 16 + col];
#pragma unroll
    for (int o4 = 0; o4 < 4; ++o4) {
      const int oct = w * 4 + o4;
      const v16h a = *(const v16h*)(wout_frag + (((size_t)oct * 16 + kc) * 32 + lane) * 16);
#pragma unroll
      for (int nt = 0; nt < 4; ++nt)
        acc[o4][nt] = wmma16(a, bfrag[nt], acc[o4][nt]);
    }
  }
#pragma unroll
  for (int o4 = 0; o4 < 4; ++o4)
#pragma unroll
    for (int nt = 0; nt < 4; ++nt)
#pragma unroll
      for (int r = 0; r < 8; ++r) {
        const int oc = (w * 4 + o4) * 16 + r + hi * 8;
        out[(size_t)oc * NSPAT + n0 + nt * 16 + col] = acc[o4][nt][r];
      }
}

extern "C" void kernel_launch(void* const* d_in, const int* in_sizes, int n_in,
                              void* d_out, int out_size, void* d_ws, size_t ws_size,
                              hipStream_t stream) {
  const float* x     = (const float*)d_in[0];
  const float* w_qkv = (const float*)d_in[1];
  const float* w_out = (const float*)d_in[2];

  char* ws = (char*)d_ws;
  float*    xred      = (float*)   (ws + WS_XRED);
  _Float16* kT_frag   = (_Float16*)(ws + WS_KT);
  _Float16* v_frag    = (_Float16*)(ws + WS_V);
  _Float16* wq_frag   = (_Float16*)(ws + WS_WQ);
  _Float16* wout_frag = (_Float16*)(ws + WS_WOUT);
  _Float16* o16       = (_Float16*)(ws + WS_O16);

  float* out      = (float*)d_out;                        // [256][32768]
  float* attn_out = (float*)d_out + (size_t)DIMC * NSPAT; // [8][32768][512]

  xred_kernel   <<<512, 256, 0, stream>>>(x, xred);
  fragkv_kernel <<<2048, 256, 0, stream>>>(w_qkv, xred, kT_frag, v_frag);
  fragw_kernel  <<<1024, 256, 0, stream>>>(w_qkv, w_out, wq_frag, wout_frag);
  attn_kernel   <<<dim3(512, 8), 128, SM_ATTN, stream>>>(x, kT_frag, v_frag, wq_frag, attn_out, o16);
  outgemm_kernel<<<512, 128, 512 * OL_STRIDE * 2, stream>>>(o16, wout_frag, out);
}
